// MoEFFN_64175401337515
// MI455X (gfx1250) — matrix operand order repaired and checked by reading, not compiled, not run on hardware
//
#include <hip/hip_runtime.h>

#define HIDDEN 1024
#define INTER  4096
#define NEXP   8
#define TILE_M 32
#define IC_W   128                 // inter chunk width per fused iteration
#define XS     (HIDDEN + 8)        // padded LDS row stride (bf16) for X tile
#define HS2    (IC_W + 8)          // padded LDS row stride (bf16) for H chunk

typedef __attribute__((ext_vector_type(16))) __bf16       v16bf;
typedef __attribute__((ext_vector_type(8)))  float        v8f;
typedef __attribute__((ext_vector_type(4)))  unsigned int v4u;
typedef __attribute__((ext_vector_type(4)))  int          v4i;

union Frag { v4u q[2]; v16bf bf; };

static __device__ __forceinline__ unsigned short f2bf(float f) {
    unsigned int u = __float_as_uint(f);
    u += 0x7FFFu + ((u >> 16) & 1u);   // round-to-nearest-even
    return (unsigned short)(u >> 16);
}

static __device__ __forceinline__ float gelu_exact(float v) {
    return 0.5f * v * (1.0f + erff(v * 0.70710678118654752f));
}

// ---------------- init: zero output + expert counts ----------------
__global__ void moe_init_kernel(float* __restrict__ out, int* __restrict__ counts, long n) {
    long i = (long)blockIdx.x * blockDim.x + threadIdx.x;
    if (i < NEXP) counts[i] = 0;
    float4* o4 = (float4*)out;
    long n4 = n >> 2;
    long stride = (long)gridDim.x * blockDim.x;
    for (long j = i; j < n4; j += stride) o4[j] = make_float4(0.f, 0.f, 0.f, 0.f);
}

// ---------------- fp32 -> bf16 row-major conversion (for X) ----------------
__global__ void moe_cvt_kernel(const float* __restrict__ src, unsigned short* __restrict__ dst, long n) {
    long stride = (long)gridDim.x * blockDim.x;
    long n4 = n >> 2;
    const float4* s4 = (const float4*)src;
    for (long j = (long)blockIdx.x * blockDim.x + threadIdx.x; j < n4; j += stride) {
        float4 f = s4[j];
        uint2 o;
        o.x = (unsigned)f2bf(f.x) | ((unsigned)f2bf(f.y) << 16);
        o.y = (unsigned)f2bf(f.z) | ((unsigned)f2bf(f.w) << 16);
        *(uint2*)(dst + j * 4) = o;
    }
}

// ------- fp32 [Rg][C] -> bf16 WMMA-B-fragment-packed [Rg/16][C/32][lane32][16] -------
// Fragment (rt,kt): lane L holds row rt*16+(L&15), cols kt*32+(L>>4)*16 .. +15.
// A whole B fragment is then 1024 contiguous bytes -> coalesced wave loads in the GEMM.
__global__ void moe_pack_kernel(const float* __restrict__ src, unsigned short* __restrict__ dst,
                                long Rg, long C) {
    long total = (Rg / 16) * (C / 32) * 32;      // one 16-elem lane-slot per work item
    long stride = (long)gridDim.x * blockDim.x;
    long kts = C / 32;
    for (long f = (long)blockIdx.x * blockDim.x + threadIdx.x; f < total; f += stride) {
        long lane = f & 31;
        long kt   = (f >> 5) % kts;
        long rt   = (f >> 5) / kts;
        const float* s = src + (rt * 16 + (lane & 15)) * C + kt * 32 + (lane >> 4) * 16;
        const float4* s4 = (const float4*)s;
        unsigned short* d = dst + f * 16;
        v4u o0, o1;
        float4 f0 = s4[0], f1 = s4[1], f2 = s4[2], f3 = s4[3];
        o0[0] = (unsigned)f2bf(f0.x) | ((unsigned)f2bf(f0.y) << 16);
        o0[1] = (unsigned)f2bf(f0.z) | ((unsigned)f2bf(f0.w) << 16);
        o0[2] = (unsigned)f2bf(f1.x) | ((unsigned)f2bf(f1.y) << 16);
        o0[3] = (unsigned)f2bf(f1.z) | ((unsigned)f2bf(f1.w) << 16);
        o1[0] = (unsigned)f2bf(f2.x) | ((unsigned)f2bf(f2.y) << 16);
        o1[1] = (unsigned)f2bf(f2.z) | ((unsigned)f2bf(f2.w) << 16);
        o1[2] = (unsigned)f2bf(f3.x) | ((unsigned)f2bf(f3.y) << 16);
        o1[3] = (unsigned)f2bf(f3.z) | ((unsigned)f2bf(f3.w) << 16);
        *(v4u*)(d)     = o0;
        *(v4u*)(d + 8) = o1;
    }
}

// ---------------- gating: one wave32 per token ----------------
__global__ void moe_gate_kernel(const float* __restrict__ x, const float* __restrict__ gw,
                                int* __restrict__ counts, int* __restrict__ tok_idx,
                                float* __restrict__ tok_w, int T) {
    __shared__ float sgw[NEXP * HIDDEN];
    for (int i = threadIdx.x; i < NEXP * HIDDEN; i += blockDim.x) sgw[i] = gw[i];
    __syncthreads();

    const int lane = threadIdx.x & 31;
    const int wv   = threadIdx.x >> 5;
    const int t    = blockIdx.x * 8 + wv;
    if (t >= T) return;

    float acc[NEXP];
#pragma unroll
    for (int e = 0; e < NEXP; ++e) acc[e] = 0.f;

    const float* xr = x + (size_t)t * HIDDEN;
#pragma unroll 4
    for (int i = 0; i < HIDDEN / 32; ++i) {
        float xv = xr[i * 32 + lane];
#pragma unroll
        for (int e = 0; e < NEXP; ++e) acc[e] += xv * sgw[e * HIDDEN + i * 32 + lane];
    }
#pragma unroll
    for (int e = 0; e < NEXP; ++e) {
#pragma unroll
        for (int off = 16; off > 0; off >>= 1) acc[e] += __shfl_xor(acc[e], off, 32);
    }
    if (lane == 0) {
        int e0 = 0; float v0 = acc[0];
#pragma unroll
        for (int e = 1; e < NEXP; ++e) if (acc[e] > v0) { v0 = acc[e]; e0 = e; }
        int e1 = -1; float v1 = -3.4e38f;
#pragma unroll
        for (int e = 0; e < NEXP; ++e) if (e != e0 && acc[e] > v1) { v1 = acc[e]; e1 = e; }
        float w0 = 1.f / (1.f + __expf(v1 - v0));   // softmax over the two selected
        float w1 = 1.f - w0;
        int p0 = atomicAdd(&counts[e0], 1);
        tok_idx[(size_t)e0 * T + p0] = t;  tok_w[(size_t)e0 * T + p0] = w0;
        int p1 = atomicAdd(&counts[e1], 1);
        tok_idx[(size_t)e1 * T + p1] = t;  tok_w[(size_t)e1 * T + p1] = w1;
    }
}

// ---------------- fused expert FFN: gelu(X W1^T + b1) W2^T + b2 ----------------
// grid = (T/TILE_M, NEXP), block = 256 (8 waves). One expert, 32 gathered tokens/block.
// Phase 1: 128-wide inter chunk, wave owns (mt, ntp) -> 2 n-tiles share one A frag.
// Phase 2: wave owns 2m x 8n persistent C tiles of the 32x1024 output.
__global__ __launch_bounds__(256, 1)
void moe_ffn_kernel(const unsigned short* __restrict__ W1p,
                    const unsigned short* __restrict__ W2p,
                    const unsigned short* __restrict__ Xb,
                    const float* __restrict__ gb1, const float* __restrict__ gb2,
                    const int* __restrict__ counts, const int* __restrict__ tok_idx,
                    const float* __restrict__ tok_w,
                    float* __restrict__ out, int T) {
    const int e    = blockIdx.y;
    const int cnt  = counts[e];
    const int m0   = blockIdx.x * TILE_M;
    if (m0 >= cnt) return;
    const int rows = min(TILE_M, cnt - m0);

    extern __shared__ unsigned char smem[];
    unsigned short* sX   = (unsigned short*)smem;            // [TILE_M][XS]
    unsigned short* sH   = sX + TILE_M * XS;                 // [TILE_M][HS2]
    int*            stok = (int*)(sH + TILE_M * HS2);        // [TILE_M]
    float*          stw  = (float*)(stok + TILE_M);          // [TILE_M]

    const int tid  = threadIdx.x;
    const int wv   = tid >> 5;
    const int lane = tid & 31;
    const int hs   = lane >> 4;     // half-wave select per WMMA 16-bit layout
    const int l15  = lane & 15;

    if (tid < TILE_M) {
        int r = (tid < rows) ? (m0 + tid) : m0;          // pad with row 0, weight 0
        stok[tid] = tok_idx[(size_t)e * T + r];
        stw[tid]  = (tid < rows) ? tok_w[(size_t)e * T + r] : 0.f;
    }
    __syncthreads();

    // gather 32 token rows of bf16 X into LDS (8 threads per row, 16B chunks)
    {
        const int row = tid >> 3;
        const unsigned short* gsrc = Xb + (size_t)stok[row] * HIDDEN;
        unsigned short* drow = sX + row * XS;
#if defined(__gfx1250__) && __has_builtin(__builtin_amdgcn_global_load_async_to_lds_b128)
#pragma unroll
        for (int i = 0; i < 16; ++i) {
            int c = (tid & 7) * 8 + i * 64;
            __builtin_amdgcn_global_load_async_to_lds_b128(
                (v4i*)(gsrc + c), (v4i*)(drow + c), 0, 0);
        }
#if __has_builtin(__builtin_amdgcn_s_wait_asynccnt)
        __builtin_amdgcn_s_wait_asynccnt(0);
#else
        asm volatile("s_wait_asynccnt 0" ::: "memory");
#endif
#else
#pragma unroll
        for (int i = 0; i < 16; ++i) {
            int c = (tid & 7) * 8 + i * 64;
            *(v4u*)(drow + c) = *(const v4u*)(gsrc + c);
        }
#endif
    }

    // fragment-packed weight bases (lane offset folded in)
    const unsigned short* pw1 =
        W1p + (size_t)e * (INTER / 16) * (HIDDEN / 32) * 512 + lane * 16;
    const unsigned short* pw2 =
        W2p + (size_t)e * (HIDDEN / 16) * (INTER / 32) * 512
            + (size_t)(wv * 8) * (INTER / 32) * 512 + lane * 16;

    // persistent phase-2 accumulators
    v8f acc2[2][8];
#pragma unroll
    for (int m = 0; m < 2; ++m)
#pragma unroll
        for (int n = 0; n < 8; ++n) acc2[m][n] = (v8f){0.f,0.f,0.f,0.f,0.f,0.f,0.f,0.f};

    const int mt  = wv & 1;                 // phase-1 M tile
    const int ntp = wv >> 1;                // phase-1 n-tile pair id (0..3)
    const unsigned short* arow = sX + (mt * 16 + l15) * XS;

    for (int ic0 = 0; ic0 < INTER; ic0 += IC_W) {
        // ---- phase 1: Hc[32x128] = gelu(X @ W1[ic0:ic0+128]^T + b1) ----
        const int it0 = (ic0 >> 4) + ntp;                // inter row-tile (first of pair)
        const unsigned short* bb0 = pw1 + (size_t)it0 * (HIDDEN / 32) * 512;
        const unsigned short* bb1 = pw1 + (size_t)(it0 + 4) * (HIDDEN / 32) * 512;
        v8f c1[2];
        c1[0] = (v8f){0.f,0.f,0.f,0.f,0.f,0.f,0.f,0.f};
        c1[1] = (v8f){0.f,0.f,0.f,0.f,0.f,0.f,0.f,0.f};

        Frag a[2], b0[2], b1[2];
        a[0].q[0]  = *(const v4u*)(arow + hs * 16);
        a[0].q[1]  = *(const v4u*)(arow + hs * 16 + 8);
        b0[0].q[0] = *(const v4u*)(bb0);
        b0[0].q[1] = *(const v4u*)(bb0 + 8);
        b1[0].q[0] = *(const v4u*)(bb1);
        b1[0].q[1] = *(const v4u*)(bb1 + 8);
#pragma unroll 4
        for (int k32 = 0; k32 < HIDDEN / 32; ++k32) {
            const int cur = k32 & 1, nxt = cur ^ 1;
            if (k32 + 1 < HIDDEN / 32) {                 // software pipeline next frags
                int kb = (k32 + 1) * 32;
                a[nxt].q[0]  = *(const v4u*)(arow + kb + hs * 16);
                a[nxt].q[1]  = *(const v4u*)(arow + kb + hs * 16 + 8);
                b0[nxt].q[0] = *(const v4u*)(bb0 + (size_t)(k32 + 1) * 512);
                b0[nxt].q[1] = *(const v4u*)(bb0 + (size_t)(k32 + 1) * 512 + 8);
                b1[nxt].q[0] = *(const v4u*)(bb1 + (size_t)(k32 + 1) * 512);
                b1[nxt].q[1] = *(const v4u*)(bb1 + (size_t)(k32 + 1) * 512 + 8);
            }
            c1[0] = __builtin_amdgcn_wmma_f32_16x16x32_bf16(false, a[cur].bf, false, b0[cur].bf,
                                                            (short)0, c1[0], false, false);
            c1[1] = __builtin_amdgcn_wmma_f32_16x16x32_bf16(false, a[cur].bf, false, b1[cur].bf,
                                                            (short)0, c1[1], false, false);
        }
        if (ic0 + IC_W < INTER)          // warm L2/WGP$ for next chunk's first W1 fragment
            __builtin_prefetch(pw1 + (size_t)(it0 + 8) * (HIDDEN / 32) * 512, 0, 0);

        const int ncol0 = ic0 + ntp * 16 + l15;
        float bias1a = gb1[(size_t)e * INTER + ncol0];
        float bias1b = gb1[(size_t)e * INTER + ncol0 + 64];
        __syncthreads();                 // prior phase-2 reads of sH are complete
#pragma unroll
        for (int i = 0; i < 8; ++i) {
            int m = mt * 16 + hs * 8 + i;
            sH[m * HS2 + ntp * 16 + l15]      = f2bf(gelu_exact(c1[0][i] + bias1a));
            sH[m * HS2 + 64 + ntp * 16 + l15] = f2bf(gelu_exact(c1[1][i] + bias1b));
        }
        __syncthreads();                 // sH chunk ready

        // ---- phase 2: Y += Hc @ W2[:, ic0:ic0+128]^T ----
#pragma unroll
        for (int kk = 0; kk < IC_W / 32; ++kk) {
            Frag a0, a1;
            const unsigned short* h0 = sH + l15 * HS2 + kk * 32;
            const unsigned short* h1 = sH + (16 + l15) * HS2 + kk * 32;
            a0.q[0] = *(const v4u*)(h0 + hs * 16);
            a0.q[1] = *(const v4u*)(h0 + hs * 16 + 8);
            a1.q[0] = *(const v4u*)(h1 + hs * 16);
            a1.q[1] = *(const v4u*)(h1 + hs * 16 + 8);

            const size_t kf = (size_t)(ic0 >> 5) + kk;   // k fragment index into packed W2
            Frag bb[2];
            bb[0].q[0] = *(const v4u*)(pw2 + kf * 512);
            bb[0].q[1] = *(const v4u*)(pw2 + kf * 512 + 8);
#pragma unroll
            for (int n2t = 0; n2t < 8; ++n2t) {
                const int cur = n2t & 1, nxt = cur ^ 1;
                if (n2t + 1 < 8) {                       // pipeline next B fragment
                    const unsigned short* nb =
                        pw2 + ((size_t)(n2t + 1) * (INTER / 32) + kf) * 512;
                    bb[nxt].q[0] = *(const v4u*)(nb);
                    bb[nxt].q[1] = *(const v4u*)(nb + 8);
                }
                acc2[0][n2t] = __builtin_amdgcn_wmma_f32_16x16x32_bf16(false, a0.bf, false, bb[cur].bf,
                                                                      (short)0, acc2[0][n2t], false, false);
                acc2[1][n2t] = __builtin_amdgcn_wmma_f32_16x16x32_bf16(false, a1.bf, false, bb[cur].bf,
                                                                      (short)0, acc2[1][n2t], false, false);
            }
        }
    }

    // ---- finalize: out[tok] += w * (Y + b2) ----
#pragma unroll
    for (int m2 = 0; m2 < 2; ++m2) {
#pragma unroll
        for (int n2t = 0; n2t < 8; ++n2t) {
            int n2 = wv * 128 + n2t * 16 + l15;
            float bias2 = gb2[(size_t)e * HIDDEN + n2];
#pragma unroll
            for (int i = 0; i < 8; ++i) {
                int m = m2 * 16 + hs * 8 + i;
                int t = stok[m];
                float wt = stw[m];
                atomicAdd(out + (size_t)t * HIDDEN + n2, wt * (acc2[m2][n2t][i] + bias2));
            }
        }
    }
}

extern "C" void kernel_launch(void* const* d_in, const int* in_sizes, int n_in,
                              void* d_out, int out_size, void* d_ws, size_t ws_size,
                              hipStream_t stream) {
    (void)n_in; (void)out_size; (void)ws_size;
    const float* x   = (const float*)d_in[0];   // [B,S,H] f32
    const float* gw  = (const float*)d_in[1];   // [E,H]
    const float* W1  = (const float*)d_in[2];   // [E,I,H]
    const float* b1  = (const float*)d_in[3];   // [E,I]
    const float* W2  = (const float*)d_in[4];   // [E,H,I]
    const float* b2  = (const float*)d_in[5];   // [E,H]
    float* out = (float*)d_out;

    const int  T  = in_sizes[0] / HIDDEN;       // 8192 tokens
    const long nW = (long)NEXP * INTER * HIDDEN;
    const long nX = (long)T * HIDDEN;

    // workspace: packed bf16 weights + bf16 activations + routing metadata
    unsigned char* ws = (unsigned char*)d_ws;
    unsigned short* W1p  = (unsigned short*)(ws);
    unsigned short* W2p  = (unsigned short*)(ws + nW * 2);
    unsigned short* Xb   = (unsigned short*)(ws + nW * 4);
    int*            cnts = (int*)  (ws + nW * 4 + nX * 2);
    int*            tidx = (int*)  (ws + nW * 4 + nX * 2 + 256);
    float*          twgt = (float*)(ws + nW * 4 + nX * 2 + 256 + (long)NEXP * T * 4);

    moe_init_kernel<<<2048, 256, 0, stream>>>(out, cnts, nX);
    moe_pack_kernel<<<2048, 256, 0, stream>>>(W1, W1p, (long)NEXP * INTER, HIDDEN);
    moe_pack_kernel<<<2048, 256, 0, stream>>>(W2, W2p, (long)NEXP * HIDDEN, INTER);
    moe_cvt_kernel <<<1024, 256, 0, stream>>>(x,  Xb,  nX);
    moe_gate_kernel<<<T / 8, 256, 0, stream>>>(x, gw, cnts, tidx, twgt, T);

    const size_t smem = (size_t)TILE_M * XS * 2 + (size_t)TILE_M * HS2 * 2
                      + TILE_M * sizeof(int) + TILE_M * sizeof(float);
    dim3 grid(T / TILE_M, NEXP);
    moe_ffn_kernel<<<grid, 256, smem, stream>>>(W1p, W2p, Xb, b1, b2,
                                                cnts, tidx, twgt, out, T);
}
